// EncoderLSTM_86474871538013
// MI455X (gfx1250) — matrix operand order repaired and checked by reading, not compiled, not run on hardware
//
#include <hip/hip_runtime.h>

typedef __bf16 bf16;
typedef __attribute__((ext_vector_type(16))) __bf16 v16bf;
typedef __attribute__((ext_vector_type(8)))  float  v8f;

#define T_STEPS 512
#define B_SZ    64
#define D_SZ    1024
#define H_SZ    1024
#define N4H     4096    // 4*H
#define NTILES  256     // 4H / 16
#define FRAG    512     // elements per fragment tile: 32 lanes * 16 bf16
#define NBLK_REC 32     // persistent-kernel workgroups
#define LDS_B_BYTES (256 * 1024)   // 2 strips * 32 kb * 4 q * 512 elem * 2B

// K index inside a 16x32 bf16 A-fragment for (lane, element j) per CDNA5 ISA:
// lanes 0-15: v0..3 -> K=0..7, v4..7 -> K=16..23 ; lanes 16-31: +8
__device__ __forceinline__ int frag_k_of(int lane, int j) {
    int v = j >> 1, odd = j & 1;
    int kbase = (lane >> 4) << 3;
    int kk = (v < 4) ? (v << 1) : (16 + ((v - 4) << 1));
    return kbase + kk + odd;
}

// Inverse: packed A-fragment flat index for matrix element (row b, col)
__device__ __forceinline__ int hp_index(int b, int col) {
    int mtile = b >> 4, mr = b & 15;
    int kb = col >> 5, ko = col & 31;
    int g = (ko >> 3) & 1;
    int lane = mr + (g << 4);
    int rem = ko - (g << 3);               // in [0,8) U [16,24)
    int odd = rem & 1;
    int kkb = rem & ~1;
    int v = (kkb < 8) ? (kkb >> 1) : (4 + ((kkb - 16) >> 1));
    int j = (v << 1) | odd;
    return (((mtile << 5) + kb) << 5 | lane) * 16 + j;
}

// ---- shared pipelined K-loop: acc[0..3] += A(16x1024) x B(1024x64 strip) ----
// ap: packed A frags for this m-tile (+lane*16), kb stride = FRAG (global)
// bp: packed B frags for this n-strip (+lane*16); strides passed in (global or LDS)
__device__ __forceinline__ void wmma_kloop(const bf16* ap, const bf16* bp,
                                           int bKbStride, v8f acc[4]) {
    v16bf a0, a1, b00, b01, b02, b03, b10, b11, b12, b13;
    a0  = *(const v16bf*)(ap);
    b00 = *(const v16bf*)(bp + 0 * FRAG);
    b01 = *(const v16bf*)(bp + 1 * FRAG);
    b02 = *(const v16bf*)(bp + 2 * FRAG);
    b03 = *(const v16bf*)(bp + 3 * FRAG);
    for (int kb = 0; kb < 32; kb += 2) {
        { // prefetch K-block kb+1 while computing kb
            const bf16* an = ap + (size_t)(kb + 1) * FRAG;
            const bf16* bn = bp + (size_t)(kb + 1) * bKbStride;
            a1  = *(const v16bf*)(an);
            b10 = *(const v16bf*)(bn + 0 * FRAG);
            b11 = *(const v16bf*)(bn + 1 * FRAG);
            b12 = *(const v16bf*)(bn + 2 * FRAG);
            b13 = *(const v16bf*)(bn + 3 * FRAG);
        }
        acc[0] = __builtin_amdgcn_wmma_f32_16x16x32_bf16(false, a0, false, b00, (short)0, acc[0], false, false);
        acc[1] = __builtin_amdgcn_wmma_f32_16x16x32_bf16(false, a0, false, b01, (short)0, acc[1], false, false);
        acc[2] = __builtin_amdgcn_wmma_f32_16x16x32_bf16(false, a0, false, b02, (short)0, acc[2], false, false);
        acc[3] = __builtin_amdgcn_wmma_f32_16x16x32_bf16(false, a0, false, b03, (short)0, acc[3], false, false);
        if (kb + 2 < 32) { // prefetch K-block kb+2 while computing kb+1
            const bf16* an = ap + (size_t)(kb + 2) * FRAG;
            const bf16* bn = bp + (size_t)(kb + 2) * bKbStride;
            a0  = *(const v16bf*)(an);
            b00 = *(const v16bf*)(bn + 0 * FRAG);
            b01 = *(const v16bf*)(bn + 1 * FRAG);
            b02 = *(const v16bf*)(bn + 2 * FRAG);
            b03 = *(const v16bf*)(bn + 3 * FRAG);
        }
        acc[0] = __builtin_amdgcn_wmma_f32_16x16x32_bf16(false, a1, false, b10, (short)0, acc[0], false, false);
        acc[1] = __builtin_amdgcn_wmma_f32_16x16x32_bf16(false, a1, false, b11, (short)0, acc[1], false, false);
        acc[2] = __builtin_amdgcn_wmma_f32_16x16x32_bf16(false, a1, false, b12, (short)0, acc[2], false, false);
        acc[3] = __builtin_amdgcn_wmma_f32_16x16x32_bf16(false, a1, false, b13, (short)0, acc[3], false, false);
    }
}

// ---------- Phase 0: conversions / packing ----------

__global__ __launch_bounds__(256) void k_pack_x(const float* __restrict__ X,
                                                bf16* __restrict__ XP) {
    long p = (long)blockIdx.x * 256 + threadIdx.x;
    int j    = (int)(p & 15);
    int lane = (int)((p >> 4) & 31);
    int kb   = (int)((p >> 9) & 31);
    long mt  = p >> 14;
    long src = (mt * 16 + (lane & 15)) * (long)D_SZ + kb * 32 + frag_k_of(lane, j);
    XP[p] = (bf16)X[src];
}

__global__ __launch_bounds__(256) void k_pack_w(const float* __restrict__ W,
                                                bf16* __restrict__ WP) {
    long p = (long)blockIdx.x * 256 + threadIdx.x;
    int j    = (int)(p & 15);
    int lane = (int)((p >> 4) & 31);
    int nb   = (int)((p >> 9) & 255);
    int kb   = (int)(p >> 17);
    long src = (long)(kb * 32 + frag_k_of(lane, j)) * N4H + nb * 16 + (lane & 15);
    WP[p] = (bf16)W[src];
}

__global__ __launch_bounds__(256) void k_init_state(const float* __restrict__ h0,
                                                    const float* __restrict__ c0,
                                                    bf16* __restrict__ HP,
                                                    float* __restrict__ Cst,
                                                    unsigned* __restrict__ bar) {
    int idx = blockIdx.x * 256 + threadIdx.x;   // < B*H
    int b = idx >> 10, col = idx & 1023;
    HP[hp_index(b, col)] = (bf16)h0[idx];
    Cst[idx] = c0[idx];
    if (idx == 0) *bar = 0u;
}

// ---------- Phase 1: Zpre = X @ Wi + b (fully parallel) ----------
__global__ __launch_bounds__(256) void k_gemm_pre(const bf16* __restrict__ XP,
                                                  const bf16* __restrict__ WiP,
                                                  const float* __restrict__ bias,
                                                  float* __restrict__ Zpre) {
    const int lane = threadIdx.x & 31;
    const int wave = threadIdx.x >> 5;
    const int gw   = blockIdx.x * 8 + wave;
    const int mt   = gw >> 6;
    const int ns   = gw & 63;
    const int n0   = ns * 64;
    const int mr   = lane & 15;
    const int hi   = lane >> 4;

    v8f acc[4];
#pragma unroll
    for (int q = 0; q < 4; ++q) {
        float bb = bias[n0 + q * 16 + mr];
#pragma unroll
        for (int r = 0; r < 8; ++r) acc[q][r] = bb;
    }

    const bf16* ap = XP + (size_t)mt * (32 * FRAG) + lane * 16;
    const bf16* bp = WiP + (size_t)(n0 >> 4) * FRAG + lane * 16;
    wmma_kloop(ap, bp, NTILES * FRAG, acc);

#pragma unroll
    for (int q = 0; q < 4; ++q) {
        float* out = Zpre + (size_t)(mt * 16 + hi * 8) * N4H + n0 + q * 16 + mr;
#pragma unroll
        for (int r = 0; r < 8; ++r) out[(size_t)r * N4H] = acc[q][r];
    }
}

// ---------- grid-wide barrier (monotonic counter, agent scope) ----------
__device__ __forceinline__ void grid_barrier(unsigned* bar, unsigned target) {
    __threadfence();
    __syncthreads();
    if (threadIdx.x == 0) {
        __hip_atomic_fetch_add(bar, 1u, __ATOMIC_ACQ_REL, __HIP_MEMORY_SCOPE_AGENT);
        while (__hip_atomic_load(bar, __ATOMIC_ACQUIRE, __HIP_MEMORY_SCOPE_AGENT) < target) {
            __builtin_amdgcn_s_sleep(2);
        }
    }
    __syncthreads();
    __threadfence();
}

// ---------- Phase 2: persistent recurrent kernel ----------
// Wave mapping: block owns 2 n-strips; wave w -> strip_local = w>>2, mt = w&3.
// Block's Wh slice (2 strips x 4 nb x 32 kb = 256 KB) staged once in LDS via
// gfx1250 async global->LDS loads, then read at LDS latency for all 512 steps.
__global__ __launch_bounds__(256) void k_recur(const bf16* __restrict__ WhP,
                                               const float* __restrict__ Zpre,
                                               float* __restrict__ Zbuf,
                                               float* __restrict__ Cst,
                                               bf16* HP,
                                               float* __restrict__ out,
                                               unsigned* bar) {
    extern __shared__ __align__(16) bf16 Bs[];   // 2*32*4*512 bf16 = 256 KB

    const int lane = threadIdx.x & 31;
    const int wave = threadIdx.x >> 5;
    const int sl   = wave >> 2;               // strip within block: 0..1
    const int mt   = wave & 3;                // m-tile: 0..3
    const int ns   = blockIdx.x * 2 + sl;     // n-strip: 0..63
    const int n0   = ns * 64;
    const int mr   = lane & 15;
    const int hi   = lane >> 4;
    const int tid  = blockIdx.x * 256 + threadIdx.x;  // 0..8191

    // ---- one-time async stage of this block's Wh slice into LDS ----
    {
        const int t0 = threadIdx.x;
        for (int i = 0; i < 64; ++i) {
            int c     = i * 256 + t0;         // 16-byte chunk id, < 16384
            int jh    = c & 1;
            int lane2 = (c >> 1) & 31;
            int q     = (c >> 6) & 3;
            int kb2   = (c >> 8) & 31;
            int sl2   = (c >> 13) & 1;
            size_t src = (((size_t)kb2 * NTILES + ((size_t)blockIdx.x * 2 + sl2) * 4 + q) * 32
                          + lane2) * 16 + jh * 8;
            size_t dst = (((size_t)(sl2 * 32 + kb2) * 4 + q) * 512 + (size_t)lane2 * 16 + jh * 8);
            unsigned lds_off = (unsigned)(size_t)(Bs + dst);          // LDS byte addr (low 32b)
            unsigned long long ga = (unsigned long long)(WhP + src);  // global byte addr
            asm volatile("global_load_async_to_lds_b128 %0, %1, off"
                         :: "v"(lds_off), "v"(ga) : "memory");
        }
        asm volatile("s_wait_asynccnt 0x0" ::: "memory");
        __syncthreads();
    }

    const bf16* ap = HP + (size_t)mt * (32 * FRAG) + lane * 16;
    const bf16* bp = Bs + (size_t)sl * (32 * 4 * FRAG) + lane * 16;  // LDS

    unsigned target = 0;
    for (int t = 0; t < T_STEPS; ++t) {
        // ---- z = Zpre[t] + h @ Wh ----
        v8f acc[4];
        const float* zp = Zpre + (size_t)t * (B_SZ * N4H)
                        + (size_t)(mt * 16 + hi * 8) * N4H + n0 + mr;
#pragma unroll
        for (int q = 0; q < 4; ++q)
#pragma unroll
            for (int r = 0; r < 8; ++r) acc[q][r] = zp[q * 16 + (size_t)r * N4H];

        wmma_kloop(ap, bp, 4 * FRAG, acc);   // B from LDS: ds_load_b128

        float* zo = Zbuf + (size_t)(mt * 16 + hi * 8) * N4H + n0 + mr;
#pragma unroll
        for (int q = 0; q < 4; ++q)
#pragma unroll
            for (int r = 0; r < 8; ++r) zo[q * 16 + (size_t)r * N4H] = acc[q][r];

        target += NBLK_REC;
        grid_barrier(bar, target);

        // ---- LSTM cell: 8192 threads x 8 elements ----
        float* ot = out + (size_t)t * (B_SZ * H_SZ);
#pragma unroll
        for (int e = 0; e < 8; ++e) {
            int idx = tid + e * 8192;
            int b = idx >> 10, col = idx & 1023;
            const float* zr = Zbuf + (size_t)b * N4H;
            float zi  = zr[col];
            float zf  = zr[H_SZ + col];
            float zg  = zr[2 * H_SZ + col];
            float zo2 = zr[3 * H_SZ + col];
            float ig = 1.0f / (1.0f + __expf(-zi));
            float fg = 1.0f / (1.0f + __expf(-zf));
            float gg = tanhf(zg);
            float og = 1.0f / (1.0f + __expf(-zo2));
            float cn = fg * Cst[idx] + ig * gg;
            float h  = og * tanhf(cn);
            Cst[idx] = cn;
            ot[idx]  = h;
            HP[hp_index(b, col)] = (bf16)h;   // next step's A fragments
        }

        target += NBLK_REC;
        grid_barrier(bar, target);
    }
}

extern "C" void kernel_launch(void* const* d_in, const int* in_sizes, int n_in,
                              void* d_out, int out_size, void* d_ws, size_t ws_size,
                              hipStream_t stream) {
    (void)in_sizes; (void)n_in; (void)out_size; (void)ws_size;
    const float* X  = (const float*)d_in[0];   // [T, B, D]
    const float* h0 = (const float*)d_in[1];   // [B, H]
    const float* c0 = (const float*)d_in[2];   // [B, H]
    const float* Wi = (const float*)d_in[3];   // [D, 4H]
    const float* Wh = (const float*)d_in[4];   // [H, 4H]
    const float* bb = (const float*)d_in[5];   // [4H]
    float* out = (float*)d_out;                // [T, B, H]

    char* ws = (char*)d_ws;
    size_t off = 0;
    bf16*  XP   = (bf16*)(ws + off);  off += (size_t)T_STEPS * B_SZ * D_SZ * 2;   // 64 MB
    bf16*  WiP  = (bf16*)(ws + off);  off += (size_t)D_SZ * N4H * 2;              // 8 MB
    bf16*  WhP  = (bf16*)(ws + off);  off += (size_t)H_SZ * N4H * 2;              // 8 MB
    float* Zpre = (float*)(ws + off); off += (size_t)T_STEPS * B_SZ * N4H * 4;    // 512 MB
    float* Zbuf = (float*)(ws + off); off += (size_t)B_SZ * N4H * 4;              // 1 MB
    bf16*  HP   = (bf16*)(ws + off);  off += (size_t)B_SZ * H_SZ * 2;             // 128 KB
    float* Cst  = (float*)(ws + off); off += (size_t)B_SZ * H_SZ * 4;             // 256 KB
    unsigned* bar = (unsigned*)(ws + off); off += 256;

    // Allow 256 KB dynamic LDS for the persistent kernel (host-side, capture-safe)
    hipFuncSetAttribute(reinterpret_cast<const void*>(k_recur),
                        hipFuncAttributeMaxDynamicSharedMemorySize, LDS_B_BYTES);

    // Phase 0: convert + pack into WMMA fragment layouts
    k_pack_x<<<(T_STEPS * B_SZ * D_SZ) / 256, 256, 0, stream>>>(X, XP);
    k_pack_w<<<(D_SZ * N4H) / 256, 256, 0, stream>>>(Wi, WiP);
    k_pack_w<<<(H_SZ * N4H) / 256, 256, 0, stream>>>(Wh, WhP);
    k_init_state<<<(B_SZ * H_SZ) / 256, 256, 0, stream>>>(h0, c0, HP, Cst, bar);

    // Phase 1: Zpre = X @ Wi + b
    k_gemm_pre<<<(2048 * 64) / 8, 256, 0, stream>>>(XP, WiP, bb, Zpre);

    // Phase 2: one persistent kernel for the whole recurrence
    k_recur<<<NBLK_REC, 256, LDS_B_BYTES, stream>>>(WhP, Zpre, Zbuf, Cst, HP, out, bar);
}
